// SelectiveSSM_90778428768286
// MI455X (gfx1250) — matrix operand order repaired and checked by reading, not compiled, not run on hardware
//
#include <hip/hip_runtime.h>

// ---------------------------------------------------------------------------
// Mamba-style selective SSM block for MI455X (gfx1250, wave32).
//  * All GEMMs: v_wmma_f32_16x16x32_bf16, bf16 operands pre-converted once
//    (removes v_cvt from the hot loop, halves operand bytes).
//  * B tiles staged in LDS via global_load_async_to_lds_b128 (ASYNCcnt),
//    double-buffered with s_wait_asynccnt + workgroup barrier, read back
//    with ds_load_b128. B traffic shared by the 8 waves of the workgroup.
//  * Scan parallelized over B*D_INNER*D_STATE lanes, wave32 shuffle reduce.
// ---------------------------------------------------------------------------

typedef __bf16 bf16;
typedef bf16  v16bf __attribute__((ext_vector_type(16)));
typedef bf16  v8bf  __attribute__((ext_vector_type(8)));
typedef float v8f   __attribute__((ext_vector_type(8)));

#define DMODEL 1024
#define DSTATE 16
#define DCONV  4
#define DINNER 2048
#define DTRANK 64
#define BSZ    2
#define LSEQ   2048
#define MROWS  (BSZ * LSEQ)   // 4096 token rows
#define NPROJ  (DTRANK + 2 * DSTATE)  // 96

// LDS tile row stride in bf16 elements (80B): 16 reading lanes map to
// distinct banks (col*20 words mod 64 all distinct), and keeps every async
// 16B chunk 16B-aligned.
#define TSTRIDE 40

// A-fragment, 16x32 bf16 (ISA 7.12.2):
//   lane<16 : row = lane,    K = {0..7, 16..23}
//   lane>=16: row = lane-16, K = {8..15, 24..31}
static __device__ __forceinline__ v16bf load_fragA(const bf16* __restrict__ p,
                                                   int ld, int lane) {
  const int row = lane & 15;
  const int cb  = (lane >> 4) << 3;              // 0 or 8
  const bf16* q = p + (long)row * ld + cb;
  v8bf lo = *(const v8bf*)q;                     // K = cb .. cb+7
  v8bf hi = *(const v8bf*)(q + 16);              // K = cb+16 .. cb+23
  return __builtin_shufflevector(lo, hi, 0,1,2,3,4,5,6,7,8,9,10,11,12,13,14,15);
}

// B-fragment from LDS tile (32x16 bf16, column-per-lane):
//   lane<16 : col = lane,    K = {0..7, 16..23};  lane>=16: col = lane-16, K = {8..15, 24..31}
static __device__ __forceinline__ v16bf load_fragB_lds(const bf16* tb, int sub,
                                                       int lane) {
  const int col = sub * 16 + (lane & 15);
  const int kb  = (lane >> 4) << 3;              // 0 or 8
  const bf16* q = tb + col * TSTRIDE + kb;
  v8bf lo = *(const v8bf*)q;                     // ds_load_b128
  v8bf hi = *(const v8bf*)(q + 16);              // ds_load_b128
  return __builtin_shufflevector(lo, hi, 0,1,2,3,4,5,6,7,8,9,10,11,12,13,14,15);
}

// C[M,N] = A[M,K] (bf16, row-major, leading dim lda) * B[N,K]^T (bf16).
// Workgroup: 8 waves -> 128 rows x (16*NT) cols. B tile staged in LDS.
// Epilogue: optional per-column bias + softplus; writes f32 and/or bf16.
template<int NT>
__global__ void gemm_bf16_wmma(const bf16* __restrict__ A,
                               const bf16* __restrict__ B,
                               float* __restrict__ Cf, bf16* __restrict__ Cb,
                               int N, int K, int lda,
                               const float* __restrict__ bias, int act) {
  constexpr int BN = 16 * NT;
  __shared__ bf16 tile[2][BN * TSTRIDE];         // double-buffered B tile
  const int lane = threadIdx.x & 31;
  const int wave = threadIdx.x >> 5;
  const int r0 = blockIdx.y * 128 + wave * 16;
  const int c0 = blockIdx.x * BN;

  // Async-copy one 32-K slab of B rows [c0, c0+BN) into tile[buf].
  // One 16B chunk per thread; BN*4 threads participate.
  // NOTE: static LDS allocation starts at offset 0 of the wave's LDS segment,
  // which is the base the async instruction's LDS address is relative to.
  auto copy_tile = [&](int buf, int k0) {
    const int t = threadIdx.x;
    if (t < BN * 4) {
      const int row = t >> 2;                    // B row (= C column) in tile
      const int ch  = (t & 3) * 8;               // bf16 element offset
      unsigned lds_off = (unsigned)(buf * BN * TSTRIDE * 2 +
                                    row * TSTRIDE * 2 + ch * 2);
      unsigned long long gaddr =
          (unsigned long long)(const void*)(B + (long)(c0 + row) * K + k0 + ch);
      asm volatile("global_load_async_to_lds_b128 %0, %1, off"
                   :: "v"(lds_off), "v"(gaddr) : "memory");
    }
  };

  v8f acc[NT] = {};
  const int KT = K / 32;
  copy_tile(0, 0);
  for (int kt = 0; kt < KT; ++kt) {
    const int k0 = kt * 32;
    asm volatile("s_wait_asynccnt 0x0" ::: "memory"); // own copies done
    __syncthreads();                                  // everyone's copies done
    if (kt + 1 < KT) copy_tile((kt + 1) & 1, k0 + 32);
    v16bf a = load_fragA(A + (long)r0 * lda + k0, lda, lane);
    if (kt + 1 < KT)
      __builtin_prefetch(A + (long)r0 * lda + k0 + 32, 0, 1);
    const bf16* tb = &tile[kt & 1][0];
#pragma unroll
    for (int t = 0; t < NT; ++t) {
      v16bf b = load_fragB_lds(tb, t, lane);
      acc[t] = __builtin_amdgcn_wmma_f32_16x16x32_bf16(
          false, a, false, b, (short)0, acc[t], false, false);
    }
  }

  // D layout (7.12.2): lane l, vgpr i -> row r0 + i + ((l>>4)<<3), col c0+(l&15)
  const int rbase = r0 + ((lane >> 4) << 3);
  const int cc    = lane & 15;
#pragma unroll
  for (int t = 0; t < NT; ++t) {
    const int col = c0 + 16 * t + cc;
    const float bv = bias ? bias[col] : 0.0f;
#pragma unroll
    for (int i = 0; i < 8; ++i) {
      float v = acc[t][i] + bv;
      if (act == 1) v = (v > 20.0f) ? v : __logf(1.0f + __expf(v));
      const long idx = (long)(rbase + i) * N + col;
      if (Cf) Cf[idx] = v;
      if (Cb) Cb[idx] = (bf16)v;
    }
  }
}

// fp32 -> bf16 bulk convert (weights / input activations, once per launch).
__global__ void cvt_bf16_kernel(const float* __restrict__ in,
                                bf16* __restrict__ out, int n) {
  const int i = blockIdx.x * blockDim.x + threadIdx.x;
  if (i < n) out[i] = (bf16)in[i];
}

// Depthwise causal conv (k=4) along L + bias + SiLU; fp32 xz in, bf16 out.
__global__ void conv_silu_kernel(const float* __restrict__ xz,
                                 const float* __restrict__ w,
                                 const float* __restrict__ bias,
                                 bf16* __restrict__ out) {
  const int idx = blockIdx.x * blockDim.x + threadIdx.x;
  if (idx >= MROWS * DINNER) return;
  const int d = idx & (DINNER - 1);
  const int m = idx >> 11;
  const int l = m & (LSEQ - 1);
  float acc = bias[d];
#pragma unroll
  for (int j = 0; j < DCONV; ++j) {
    const int ll = l - (DCONV - 1) + j;
    if (ll >= 0)
      acc += w[d * DCONV + j] * xz[(long)(m - (DCONV - 1) + j) * (2 * DINNER) + d];
  }
  out[idx] = (bf16)(acc / (1.0f + __expf(-acc)));   // SiLU
}

// Selective scan: one lane per (b, d, n); 16 consecutive lanes share one
// channel, y reduced with wave32 shuffles; writes bf16 y for the out-proj.
__global__ void scan_kernel(const bf16* __restrict__ xca,
                            const bf16* __restrict__ dt,
                            const float* __restrict__ xp,
                            const float* __restrict__ xz,
                            const float* __restrict__ A_log,
                            const float* __restrict__ Dp,
                            bf16* __restrict__ y) {
  const int t = blockIdx.x * blockDim.x + threadIdx.x;   // 65536 lanes
  const int n = t & (DSTATE - 1);
  const int d = (t >> 4) & (DINNER - 1);
  const int b = t >> 15;
  const float a  = -__expf(A_log[d * DSTATE + n]);
  const float Dd = Dp[d];
  float h = 0.0f;
  for (int l = 0; l < LSEQ; ++l) {
    const long m   = (long)b * LSEQ + l;
    const float dv = (float)dt [m * DINNER + d];
    const float xv = (float)xca[m * DINNER + d];
    const float Bv = xp[m * NPROJ + DTRANK + n];
    const float Cv = xp[m * NPROJ + DTRANK + DSTATE + n];
    h = __expf(dv * a) * h + (dv * xv) * Bv;
    float yp = h * Cv;
    yp += __shfl_xor(yp, 1, 32);
    yp += __shfl_xor(yp, 2, 32);
    yp += __shfl_xor(yp, 4, 32);
    yp += __shfl_xor(yp, 8, 32);
    if (n == 0) {
      const float zv = xz[m * (2 * DINNER) + DINNER + d];
      y[m * DINNER + d] = (bf16)((yp + xv * Dd) * (zv / (1.0f + __expf(-zv))));
    }
  }
}

extern "C" void kernel_launch(void* const* d_in, const int* in_sizes, int n_in,
                              void* d_out, int out_size, void* d_ws, size_t ws_size,
                              hipStream_t stream) {
  const float* x       = (const float*)d_in[0];  // [2,2048,1024]
  const float* W_in    = (const float*)d_in[1];  // [4096,1024]
  const float* conv_w  = (const float*)d_in[2];  // [2048,1,4]
  const float* conv_b  = (const float*)d_in[3];  // [2048]
  const float* W_xproj = (const float*)d_in[4];  // [96,2048]
  const float* W_dt    = (const float*)d_in[5];  // [2048,64]
  const float* b_dt    = (const float*)d_in[6];  // [2048]
  const float* A_log   = (const float*)d_in[7];  // [2048,16]
  const float* Dvec    = (const float*)d_in[8];  // [2048]
  const float* W_out   = (const float*)d_in[9];  // [1024,2048]
  float* out = (float*)d_out;                    // [2,2048,1024]

  // Workspace: f32 { xz 16.78M | xp 0.39M } then bf16 pool (~72.7MB). ~141MB.
  float* ws_f = (float*)d_ws;
  float* xz = ws_f;                              // [4096, 4096] f32
  float* xp = ws_f + 16777216;                   // [4096, 96]   f32
  bf16* bb     = (bf16*)(ws_f + 17170432);
  bf16* xbf    = bb + 0;                         // [4096,1024]
  bf16* Winbf  = bb + 4194304;                   // [4096,1024]
  bf16* Wxpbf  = bb + 8388608;                   // [96,2048]
  bf16* Wdtbf  = bb + 8585216;                   // [2048,64]
  bf16* Woutbf = bb + 8716288;                   // [1024,2048]
  bf16* xcabf  = bb + 10813440;                  // [4096,2048]
  bf16* xpbf   = bb + 19202048;                  // [4096,96]
  bf16* dtbf   = bb + 19595264;                  // [4096,2048]
  bf16* ybf    = bb + 27983872;                  // [4096,2048]

  // 0) one-time bf16 conversions of x + weights
  cvt_bf16_kernel<<<(4194304 + 255) / 256, 256, 0, stream>>>(x, xbf, 4194304);
  cvt_bf16_kernel<<<(4194304 + 255) / 256, 256, 0, stream>>>(W_in, Winbf, 4194304);
  cvt_bf16_kernel<<<(196608 + 255) / 256, 256, 0, stream>>>(W_xproj, Wxpbf, 196608);
  cvt_bf16_kernel<<<(131072 + 255) / 256, 256, 0, stream>>>(W_dt, Wdtbf, 131072);
  cvt_bf16_kernel<<<(2097152 + 255) / 256, 256, 0, stream>>>(W_out, Woutbf, 2097152);

  // 1) xz = x @ W_in^T            (M=4096, N=4096, K=1024)
  gemm_bf16_wmma<4><<<dim3(4096 / 64, MROWS / 128), 256, 0, stream>>>(
      xbf, Winbf, xz, nullptr, 2 * DINNER, DMODEL, DMODEL, nullptr, 0);
  // 2) depthwise causal conv + bias + SiLU -> xca (bf16)
  conv_silu_kernel<<<(MROWS * DINNER) / 256, 256, 0, stream>>>(
      xz, conv_w, conv_b, xcabf);
  // 3) xp = xca @ W_xproj^T       (M=4096, N=96, K=2048); f32 for scan + bf16
  gemm_bf16_wmma<2><<<dim3(NPROJ / 32, MROWS / 128), 256, 0, stream>>>(
      xcabf, Wxpbf, xp, xpbf, NPROJ, DINNER, DINNER, nullptr, 0);
  // 4) dt = softplus(xp[:, :64] @ W_dt^T + b_dt)   (N=2048, K=64, lda=96)
  gemm_bf16_wmma<4><<<dim3(DINNER / 64, MROWS / 128), 256, 0, stream>>>(
      xpbf, Wdtbf, nullptr, dtbf, DINNER, DTRANK, NPROJ, b_dt, 1);
  // 5) selective scan + skip(D) + SiLU(z) gate -> ybf
  scan_kernel<<<(BSZ * DINNER * DSTATE) / 256, 256, 0, stream>>>(
      xcabf, dtbf, xp, xz, A_log, Dvec, ybf);
  // 6) out = y @ W_out^T          (M=4096, N=1024, K=2048)
  gemm_bf16_wmma<4><<<dim3(DMODEL / 64, MROWS / 128), 256, 0, stream>>>(
      ybf, Woutbf, out, nullptr, DMODEL, DINNER, DINNER, nullptr, 0);
}
